// LatentResponse_78640851190490
// MI455X (gfx1250) — compile-verified, matrix operand order checked by reading, not run on hardware
//
#include <hip/hip_runtime.h>

// ---------------------------------------------------------------------------
// Types for CDNA5 WMMA (wave32): V_WMMA_F32_16X16X32_F16
// ---------------------------------------------------------------------------
typedef __attribute__((ext_vector_type(16))) _Float16 v16h;
typedef __attribute__((ext_vector_type(8)))  _Float16 v8h;
typedef __attribute__((ext_vector_type(8)))  float    v8f;
typedef unsigned int   u32;
typedef unsigned short u16;

#define NCAND 4096
#define NANCH 8192
#define DDIM  512
#define NLVL  4

// ---------------------------------------------------------------------------
// Packed hi/lo f16 representation of an fp32 value (4 bytes, ~21-bit mantissa):
//   x ~= hi + lo,  hi = f16(x), lo = f16(x - hi)
// Stored as u32 = {lo[31:16], hi[15:0]}. Same footprint as fp32, but staging
// into WMMA operand LDS needs only bitfield splits (no float conversion math).
// ---------------------------------------------------------------------------
__device__ __forceinline__ u32 pack_f32(float x) {
  _Float16 h = (_Float16)x;
  _Float16 l = (_Float16)(x - (float)h);
  return (u32)__builtin_bit_cast(u16, h) | ((u32)__builtin_bit_cast(u16, l) << 16);
}
__device__ __forceinline__ void unpack_halves(u32 p, _Float16& h, _Float16& l) {
  h = __builtin_bit_cast(_Float16, (u16)(p & 0xffffu));
  l = __builtin_bit_cast(_Float16, (u16)(p >> 16));
}
__device__ __forceinline__ float unpack_f32(u32 p) {
  _Float16 h, l;
  unpack_halves(p, h, l);
  return (float)h + (float)l;
}

// Build a 16-half operand fragment from LDS.
// Documented 16-bit A layout (16x32): lanes 0-15 hold K {0..7, 16..23},
// lanes 16-31 hold K {8..15, 24..31}; B mirrored with lane%16 = column.
__device__ __forceinline__ v16h load_frag(const _Float16* row, int lh) {
  const _Float16* p = row + lh * 8;
  v8h lo = *(const v8h*)(p);        // K = base .. base+7
  v8h hi = *(const v8h*)(p + 16);   // K = base+16 .. base+23
  v16h f;
  ((v8h*)&f)[0] = lo;
  ((v8h*)&f)[1] = hi;
  return f;
}

#define WMMA_F16(A, B, C) \
  __builtin_amdgcn_wmma_f32_16x16x32_f16(false, (A), false, (B), (short)0, (C), false, false)

// ---------------------------------------------------------------------------
// One-time fp32 -> packed hi/lo conversion (level-0 inputs, weights).
// ---------------------------------------------------------------------------
__global__ __launch_bounds__(256)
void pack_kernel(const float* __restrict__ X, u32* __restrict__ P, int n) {
  int i = blockIdx.x * 256 + threadIdx.x;
  if (i < n) P[i] = pack_f32(X[i]);
}

// ---------------------------------------------------------------------------
// GEMM: C[rows,512] = A[rows,512] @ W[512,512], packed hi/lo in AND out.
// 3 WMMAs per k-step (hi*hi + hi*lo + lo*hi) -> ~fp32 accuracy on matrix pipe.
// Block: 128 threads (4 waves). Tile: 16 rows x 64 cols. grid=(rows/16, 8).
// ---------------------------------------------------------------------------
__global__ __launch_bounds__(128)
void gemm512_wmma(const u32* __restrict__ A, const u32* __restrict__ W,
                  u32* __restrict__ C) {
  __shared__ alignas(16) _Float16 aHi[16 * 32];
  __shared__ alignas(16) _Float16 aLo[16 * 32];
  __shared__ alignas(16) _Float16 wHi[64 * 32];
  __shared__ alignas(16) _Float16 wLo[64 * 32];

  const int tid = threadIdx.x;
  const int wv  = tid >> 5;
  const int ln  = tid & 15;
  const int lh  = (tid & 31) >> 4;
  const int rt  = blockIdx.x * 16;
  const int ct  = blockIdx.y * 64;

  v8f acc = {};
  for (int k0 = 0; k0 < DDIM; k0 += 32) {
    // Stage A tile [16 x 32]: pure load + split, no conversion math.
    for (int i = tid; i < 16 * 32; i += 128) {
      u32 p = A[(rt + (i >> 5)) * DDIM + k0 + (i & 31)];
      _Float16 h, l;
      unpack_halves(p, h, l);
      aHi[i] = h;
      aLo[i] = l;
    }
    // Stage W tile [32 x 64], transposed to [col][k] for contiguous frag reads.
    for (int i = tid; i < 32 * 64; i += 128) {
      int kk = i >> 6, nn = i & 63;
      u32 p = W[(k0 + kk) * DDIM + ct + nn];
      _Float16 h, l;
      unpack_halves(p, h, l);
      wHi[nn * 32 + kk] = h;
      wLo[nn * 32 + kk] = l;
    }
    __syncthreads();

    v16h afh = load_frag(&aHi[ln * 32], lh);
    v16h afl = load_frag(&aLo[ln * 32], lh);
    const int bc = wv * 16 + ln;
    v16h bfh = load_frag(&wHi[bc * 32], lh);
    v16h bfl = load_frag(&wLo[bc * 32], lh);

    acc = WMMA_F16(afh, bfh, acc);   // hi*hi
    acc = WMMA_F16(afh, bfl, acc);   // hi*lo
    acc = WMMA_F16(afl, bfh, acc);   // lo*hi
    __syncthreads();
  }

  // C layout: VGPR r, lane -> M = r + 8*lh, N = ln. Repack fp32 acc -> hi/lo.
#pragma unroll
  for (int r = 0; r < 8; ++r) {
    int m = r + 8 * lh;
    C[(rt + m) * DDIM + ct + wv * 16 + ln] = pack_f32(acc[r]);
  }
}

// ---------------------------------------------------------------------------
// Row squared-norms from packed responses: out[row] = sum_d (hi+lo)^2.
// Uses the same reconstructed values as the distance kernel for consistency.
// ---------------------------------------------------------------------------
__global__ __launch_bounds__(128)
void rownorm_kernel(const u32* __restrict__ X, float* __restrict__ out) {
  __shared__ float red[128];
  const int row = blockIdx.x, tid = threadIdx.x;
  float s = 0.0f;
#pragma unroll
  for (int k = 0; k < 4; ++k) {
    float v = unpack_f32(X[row * DDIM + tid + 128 * k]);
    s += v * v;
  }
  red[tid] = s;
  __syncthreads();
  for (int off = 64; off > 0; off >>= 1) {
    if (tid < off) red[tid] += red[tid + off];
    __syncthreads();
  }
  if (tid == 0) out[row] = red[0];
}

// ---------------------------------------------------------------------------
// Fused distance + running top-4 + weighted score for one level.
// Block: 128 threads (4 waves) = 16 cands vs all 8192 anchors.
// Cross term via 3-term hi/lo WMMA; sq = ||c||^2 + ||a||^2 - 2*c.a ; sqrt.
// Each lane keeps a per-row register top-4 (values only; the score needs no
// indices), merged through LDS at the end. Next anchor chunk is prefetched
// (global_prefetch_b8) while the current chunk runs through the matrix pipe.
// ---------------------------------------------------------------------------
__global__ __launch_bounds__(128)
void dist_topk_kernel(const u32* __restrict__ cresp, const u32* __restrict__ aresp,
                      const float* __restrict__ cnorm, const float* __restrict__ anorm,
                      const float* __restrict__ wts, float* __restrict__ score_out) {
  __shared__ alignas(16) _Float16 aHi[16 * DDIM];   // 16 KB
  __shared__ alignas(16) _Float16 aLo[16 * DDIM];   // 16 KB
  __shared__ alignas(16) _Float16 bHi[64 * 32];     //  4 KB
  __shared__ alignas(16) _Float16 bLo[64 * 32];     //  4 KB
  __shared__ float mergeBuf[16 * 256];              // 16 KB
  __shared__ float cn[16];

  const int tid = threadIdx.x;
  const int wv  = tid >> 5;
  const int ln  = tid & 15;
  const int lh  = (tid & 31) >> 4;
  const int ct  = blockIdx.x * 16;

  // Stage the 16 candidate rows (full K) once.
  for (int i = tid; i < 16 * DDIM; i += 128) {
    _Float16 h, l;
    unpack_halves(cresp[(ct + (i >> 9)) * DDIM + (i & 511)], h, l);
    aHi[i] = h;
    aLo[i] = l;
  }
  if (tid < 16) cn[tid] = cnorm[ct + tid];
  __syncthreads();

  float t4[8][4];
#pragma unroll
  for (int r = 0; r < 8; ++r)
#pragma unroll
    for (int j = 0; j < 4; ++j) t4[r][j] = 3.0e38f;

  for (int cb = 0; cb < NANCH; cb += 64) {
    // Latency hint: prefetch the next anchor chunk while this one computes.
    int cbn = cb + 64;
    if (cbn < NANCH)
      __builtin_prefetch(&aresp[(size_t)(cbn + (tid >> 1)) * DDIM + (tid & 1) * 256], 0, 0);

    v8f acc = {};
    for (int k0 = 0; k0 < DDIM; k0 += 32) {
      // Stage anchor k-slice [64 anchors x 32 k]; anchor rows are B columns
      // and are k-contiguous in memory, so layout is already [col][k].
      for (int i = tid; i < 64 * 32; i += 128) {
        _Float16 h, l;
        unpack_halves(aresp[(cb + (i >> 5)) * DDIM + k0 + (i & 31)], h, l);
        bHi[i] = h;
        bLo[i] = l;
      }
      __syncthreads();

      v16h afh = load_frag(&aHi[ln * DDIM + k0], lh);
      v16h afl = load_frag(&aLo[ln * DDIM + k0], lh);
      const int bc = wv * 16 + ln;
      v16h bfh = load_frag(&bHi[bc * 32], lh);
      v16h bfl = load_frag(&bLo[bc * 32], lh);

      acc = WMMA_F16(afh, bfh, acc);
      acc = WMMA_F16(afh, bfl, acc);
      acc = WMMA_F16(afl, bfh, acc);
      __syncthreads();
    }

    const float an = anorm[cb + wv * 16 + ln];
#pragma unroll
    for (int r = 0; r < 8; ++r) {
      int m = r + 8 * lh;                      // local cand row
      float sq = cn[m] + an - 2.0f * acc[r];
      float d  = sqrtf(fmaxf(sq, 1e-12f));
      if (d < t4[r][3]) {                      // ascending insertion
        t4[r][3] = d;
        if (t4[r][3] < t4[r][2]) { float x = t4[r][2]; t4[r][2] = t4[r][3]; t4[r][3] = x; }
        if (t4[r][2] < t4[r][1]) { float x = t4[r][1]; t4[r][1] = t4[r][2]; t4[r][2] = x; }
        if (t4[r][1] < t4[r][0]) { float x = t4[r][0]; t4[r][0] = t4[r][1]; t4[r][1] = x; }
      }
    }
  }

  // Merge per-lane top-4s -> global top-4 per candidate (any global top-4
  // member is necessarily inside some lane's local top-4 of its substream).
#pragma unroll
  for (int r = 0; r < 8; ++r) {
    int m = r + 8 * lh;
    int base = m * 256 + (wv * 16 + ln) * 4;
#pragma unroll
    for (int j = 0; j < 4; ++j) mergeBuf[base + j] = t4[r][j];
  }
  __syncthreads();

  if (tid < 16) {
    float b0 = 3.0e38f, b1 = 3.0e38f, b2 = 3.0e38f, b3 = 3.0e38f;
    const float* p = &mergeBuf[tid * 256];
    for (int i = 0; i < 256; ++i) {
      float d = p[i];
      if (d < b3) {
        b3 = d;
        if (b3 < b2) { float x = b2; b2 = b3; b3 = x; }
        if (b2 < b1) { float x = b1; b1 = b2; b2 = x; }
        if (b1 < b0) { float x = b0; b0 = b1; b1 = x; }
      }
    }
    // score = ascending-sorted smallest-4 dot neighbor_wts
    score_out[ct + tid] = b0 * wts[0] + b1 * wts[1] + b2 * wts[2] + b3 * wts[3];
  }
}

// ---------------------------------------------------------------------------
// gold = all(score[l+1] - score[l] < 0); picks = gold ? cands : 0. grid=4096.
// ---------------------------------------------------------------------------
__global__ __launch_bounds__(128)
void finalize_kernel(const float* __restrict__ cands, const float* __restrict__ score,
                     float* __restrict__ picks, float* __restrict__ gold) {
  const int i = blockIdx.x, tid = threadIdx.x;
  float s0 = score[i];
  float s1 = score[NCAND + i];
  float s2 = score[2 * NCAND + i];
  float s3 = score[3 * NCAND + i];
  bool g = (s1 < s0) && (s2 < s1) && (s3 < s2);
#pragma unroll
  for (int k = 0; k < 4; ++k) {
    int c = tid + 128 * k;
    picks[i * DDIM + c] = g ? cands[i * DDIM + c] : 0.0f;
  }
  if (tid == 0) gold[i] = g ? 1.0f : 0.0f;
}

// ---------------------------------------------------------------------------
// Host-side orchestration (all on `stream`, graph-capture safe, deterministic).
// ---------------------------------------------------------------------------
extern "C" void kernel_launch(void* const* d_in, const int* in_sizes, int n_in,
                              void* d_out, int out_size, void* d_ws, size_t ws_size,
                              hipStream_t stream) {
  (void)in_sizes; (void)n_in; (void)out_size; (void)ws_size;

  const float* cands   = (const float*)d_in[0];  // [4096,512]
  const float* anchors = (const float*)d_in[1];  // [8192,512]
  const float* W_enc   = (const float*)d_in[2];  // [512,512]
  const float* W_dec   = (const float*)d_in[3];  // [512,512]
  const float* wts     = (const float*)d_in[4];  // [4]

  float* out   = (float*)d_out;
  float* picks = out;                             // [4096,512]
  float* score = out + (size_t)NCAND * DDIM;      // [4,4096]
  float* gold  = score + (size_t)NLVL * NCAND;    // [4096]

  const size_t CR = (size_t)NCAND * DDIM;
  const size_t AR = (size_t)NANCH * DDIM;
  const size_t WW = (size_t)DDIM * DDIM;

  u32* wsu   = (u32*)d_ws;
  u32* crp   = wsu;                 // 4 levels * CR  (packed cand responses)
  u32* arp   = crp + 4 * CR;        // 4 levels * AR  (packed anchor responses)
  u32* tmp   = arp + 4 * AR;        // AR             (packed GEMM intermediate)
  u32* wencp = tmp + AR;            // WW
  u32* wdecp = wencp + WW;          // WW
  float* cnorm = (float*)(wdecp + WW);            // 4*NCAND
  float* anorm = cnorm + (size_t)NLVL * NCAND;    // 4*NANCH

  dim3 blk(128);

  // One-time pack of level-0 inputs and weights into hi/lo f16 form.
  pack_kernel<<<(int)(CR / 256), 256, 0, stream>>>(cands, crp, (int)CR);
  pack_kernel<<<(int)(AR / 256), 256, 0, stream>>>(anchors, arp, (int)AR);
  pack_kernel<<<(int)(WW / 256), 256, 0, stream>>>(W_enc, wencp, (int)WW);
  pack_kernel<<<(int)(WW / 256), 256, 0, stream>>>(W_dec, wdecp, (int)WW);

  // Level responses: Z_{l+1} = (Z_l @ W_dec) @ W_enc for cands and anchors.
  for (int l = 0; l < 3; ++l) {
    gemm512_wmma<<<dim3(NCAND / 16, DDIM / 64), blk, 0, stream>>>(crp + l * CR, wdecp, tmp);
    gemm512_wmma<<<dim3(NCAND / 16, DDIM / 64), blk, 0, stream>>>(tmp, wencp, crp + (l + 1) * CR);
    gemm512_wmma<<<dim3(NANCH / 16, DDIM / 64), blk, 0, stream>>>(arp + l * AR, wdecp, tmp);
    gemm512_wmma<<<dim3(NANCH / 16, DDIM / 64), blk, 0, stream>>>(tmp, wencp, arp + (l + 1) * AR);
  }

  // Squared norms + fused distance / top-4 / score per level.
  for (int l = 0; l < NLVL; ++l) {
    rownorm_kernel<<<NCAND, blk, 0, stream>>>(crp + l * CR, cnorm + (size_t)l * NCAND);
    rownorm_kernel<<<NANCH, blk, 0, stream>>>(arp + l * AR, anorm + (size_t)l * NANCH);
    dist_topk_kernel<<<NCAND / 16, blk, 0, stream>>>(
        crp + l * CR, arp + l * AR, cnorm + (size_t)l * NCAND, anorm + (size_t)l * NANCH,
        wts, score + (size_t)l * NCAND);
  }

  finalize_kernel<<<NCAND, blk, 0, stream>>>(cands, score, picks, gold);
}
